// MHSA_38019050504842
// MI455X (gfx1250) — matrix operand order repaired and checked by reading, not compiled, not run on hardware
//
#include <hip/hip_runtime.h>
#include <hip/hip_bf16.h>
#include <math.h>

// ---------------- problem constants ----------------
#define BB   2
#define TT   2048
#define CC   1024
#define HH   16
#define DD   64
#define N3   (3 * CC)     // 3072
#define MROWS (BB * TT)   // 4096

typedef __attribute__((ext_vector_type(16))) __bf16        v16bf;
typedef __attribute__((ext_vector_type(8)))  float         v8f;
typedef __attribute__((ext_vector_type(4)))  float         v4f;
typedef __attribute__((ext_vector_type(4)))  unsigned int  v4u;
typedef __attribute__((ext_vector_type(2)))  unsigned int  v2u;

union BF16x16 {
    v16bf          v;
    unsigned short us[16];
    unsigned int   u[8];
    v4u            q[2];
};

union BF16x4 {
    unsigned short us[4];
    v2u            q;
};

__device__ __forceinline__ unsigned short f2bf(float f) {
    __bf16 h = (__bf16)f;                       // native RNE f32->bf16
    return __builtin_bit_cast(unsigned short, h);
}

__device__ __forceinline__ v8f wmma_bf16(v16bf a, v16bf b, v8f c) {
    return __builtin_amdgcn_wmma_f32_16x16x32_bf16(
        false, a, false, b, (short)0, c, false, false);
}

// ===================================================================
// Pack kernels: fp32 -> bf16 once, outside the GEMM hot loops.
// ===================================================================
// x [4096,1024] row-major -> bf16 same layout. 4 elements/thread.
__global__ void cvt_x_kernel(const float* __restrict__ x,
                             unsigned short* __restrict__ xb) {
    size_t i = ((size_t)blockIdx.x * blockDim.x + threadIdx.x) * 4;
    v4f f = *(const v4f*)(x + i);
    BF16x4 o;
    o.us[0] = f2bf(f.x); o.us[1] = f2bf(f.y);
    o.us[2] = f2bf(f.z); o.us[3] = f2bf(f.w);
    *(v2u*)(xb + i) = o.q;
}

// W [K,N] row-major fp32 -> Wt [N,K] bf16 (transposed, K contiguous).
__global__ void cvt_wt_kernel(const float* __restrict__ W,
                              unsigned short* __restrict__ Wt,
                              int K, int N) {
    int t  = (int)(blockIdx.x * blockDim.x + threadIdx.x);
    int n  = t % N;
    int k0 = (t / N) * 4;
    BF16x4 o;
    #pragma unroll
    for (int i = 0; i < 4; ++i)
        o.us[i] = f2bf(W[(size_t)(k0 + i) * N + n]);
    *(v2u*)(Wt + (size_t)n * K + k0) = o.q;
}

// ===================================================================
// Kernel 1: qkv = xb @ Wqkv_t^T + b_qkv -> Q[B,H,T,D], K[B,H,T,D], Vt[B,H,D,T]
// One wave computes a 16(M) x 64(N) tile: A fragment reused across 4 WMMAs.
// All fragment loads are b128.
// ===================================================================
__global__ void qkv_gemm_kernel(const unsigned short* __restrict__ xb,
                                const unsigned short* __restrict__ Wt,  // [N3, CC]
                                const float* __restrict__ bqkv,
                                unsigned short* __restrict__ Q,
                                unsigned short* __restrict__ Kb,
                                unsigned short* __restrict__ Vt) {
    const int wave = (int)((blockIdx.x * blockDim.x + threadIdx.x) >> 5);
    const int lane = threadIdx.x & 31;
    const int tilesN = N3 / 64;                  // 48
    const int tileN = (wave % tilesN) * 64;
    const int tileM = (wave / tilesN) * 16;
    if (tileM >= MROWS) return;
    const int hg = lane >> 4;
    const int ln = lane & 15;

    v8f acc[4];
    #pragma unroll
    for (int s = 0; s < 4; ++s) acc[s] = (v8f){0.f,0.f,0.f,0.f,0.f,0.f,0.f,0.f};

    const unsigned short* arow = xb + (size_t)(tileM + ln) * CC;
    const unsigned short* brow[4];
    #pragma unroll
    for (int s = 0; s < 4; ++s)
        brow[s] = Wt + (size_t)(tileN + s * 16 + ln) * CC;

    for (int k0 = 0; k0 < CC; k0 += 32) {
        BF16x16 a;
        a.q[0] = *(const v4u*)(arow + k0 + hg * 8);        // K = hg*8 .. +7
        a.q[1] = *(const v4u*)(arow + k0 + 16 + hg * 8);   // K = 16+hg*8 .. +7
        #pragma unroll
        for (int s = 0; s < 4; ++s) {
            BF16x16 b;
            b.q[0] = *(const v4u*)(brow[s] + k0 + hg * 8);      // K = hg*8 .. +7
            b.q[1] = *(const v4u*)(brow[s] + k0 + 16 + hg * 8); // K = 16+hg*8 .. +7
            acc[s] = wmma_bf16(a.v, b.v, acc[s]);
        }
    }

    // Epilogue: add bias, scatter into Q / K / V^T (bf16).
    #pragma unroll
    for (int s = 0; s < 4; ++s) {
        int coln = tileN + s * 16 + ln;
        int sec  = coln >> 10;               // 0=Q, 1=K, 2=V
        int rem  = coln & 1023;
        int h    = rem >> 6;
        int d    = rem & 63;
        float bias = bqkv[coln];
        #pragma unroll
        for (int r = 0; r < 8; ++r) {
            int rowM = tileM + r + 8 * hg;
            int bidx = rowM >> 11;
            int t    = rowM & (TT - 1);
            unsigned short bv = f2bf(acc[s][r] + bias);
            if (sec == 0)
                Q [(((size_t)bidx * HH + h) * TT + t) * DD + d] = bv;
            else if (sec == 1)
                Kb[(((size_t)bidx * HH + h) * TT + t) * DD + d] = bv;
            else
                Vt[(((size_t)bidx * HH + h) * DD + d) * TT + t] = bv;
        }
    }
}

// ===================================================================
// Kernel 2: causal flash attention per (b,h,16-query tile). One wave per block.
// ===================================================================
__global__ void flash_attn_kernel(const unsigned short* __restrict__ Q,
                                  const unsigned short* __restrict__ K,
                                  const unsigned short* __restrict__ Vt,
                                  unsigned short* __restrict__ attn) {
    __shared__ unsigned short ldsP[16][32];      // P tile (C-layout -> A-layout)

    const int bid = blockIdx.x;
    const int qt  = bid & (TT / 16 - 1);
    const int bh  = bid >> 7;
    const int b   = bh >> 4;
    const int h   = bh & 15;
    const int lane = threadIdx.x & 31;
    const int hg = lane >> 4;
    const int ln = lane & 15;
    const int qBase = qt * 16;

    const unsigned short* Qb = Q  + ((size_t)bh * TT + qBase) * DD;
    const unsigned short* Kb = K  + (size_t)bh * TT * DD;
    const unsigned short* Vb = Vt + (size_t)bh * DD * TT;

    // Q fragments (A 16x64 -> two 16x32 frags), b128 loads.
    BF16x16 qa[2];
    {
        const unsigned short* qrow = Qb + (size_t)ln * DD;
        #pragma unroll
        for (int kc = 0; kc < 2; ++kc) {
            qa[kc].q[0] = *(const v4u*)(qrow + kc * 32 + hg * 8);
            qa[kc].q[1] = *(const v4u*)(qrow + kc * 32 + 16 + hg * 8);
        }
    }

    float m[8], l[8];
    v8f acc[4];
    #pragma unroll
    for (int r = 0; r < 8; ++r) { m[r] = -INFINITY; l[r] = 0.f; }
    #pragma unroll
    for (int dc = 0; dc < 4; ++dc) acc[dc] = (v8f){0.f,0.f,0.f,0.f,0.f,0.f,0.f,0.f};

    const int nblk = (qBase + 16 + 31) >> 5;
    for (int blk = 0; blk < nblk; ++blk) {
        const int s0 = blk * 32;

        // Prefetch next key block (global_prefetch_b8 path).
        if (blk + 1 < nblk) {
            __builtin_prefetch(Kb + (size_t)(s0 + 32 + ln) * DD, 0, 1);
            __builtin_prefetch(Kb + (size_t)(s0 + 48 + ln) * DD, 0, 1);
        }

        // ---- S = Q K^T for keys [s0, s0+32) ----
        v8f sA0 = (v8f){0.f,0.f,0.f,0.f,0.f,0.f,0.f,0.f};
        v8f sA1 = sA0;
        const unsigned short* k0p = Kb + (size_t)(s0 + ln) * DD;
        const unsigned short* k1p = Kb + (size_t)(s0 + 16 + ln) * DD;
        #pragma unroll
        for (int kc = 0; kc < 2; ++kc) {
            BF16x16 kb0, kb1;
            kb0.q[0] = *(const v4u*)(k0p + kc * 32 + hg * 8);
            kb0.q[1] = *(const v4u*)(k0p + kc * 32 + 16 + hg * 8);
            kb1.q[0] = *(const v4u*)(k1p + kc * 32 + hg * 8);
            kb1.q[1] = *(const v4u*)(k1p + kc * 32 + 16 + hg * 8);
            sA0 = wmma_bf16(qa[kc].v, kb0.v, sA0);
            sA1 = wmma_bf16(qa[kc].v, kb1.v, sA1);
        }

        // ---- online softmax update ----
        float p0[8], p1[8];
        #pragma unroll
        for (int r = 0; r < 8; ++r) {
            int qi  = qBase + r + 8 * hg;
            int k0i = s0 + ln;
            int k1i = s0 + 16 + ln;
            float sv0 = (k0i <= qi) ? sA0[r] * 0.125f : -INFINITY;
            float sv1 = (k1i <= qi) ? sA1[r] * 0.125f : -INFINITY;
            float bm = fmaxf(sv0, sv1);
            #pragma unroll
            for (int off = 1; off < 16; off <<= 1)
                bm = fmaxf(bm, __shfl_xor(bm, off, 32));
            float mn   = fmaxf(m[r], bm);           // finite: key s0 <= qi always
            float corr = __expf(m[r] - mn);
            float e0 = (k0i <= qi) ? __expf(sv0 - mn) : 0.f;
            float e1 = (k1i <= qi) ? __expf(sv1 - mn) : 0.f;
            float rs = e0 + e1;
            #pragma unroll
            for (int off = 1; off < 16; off <<= 1)
                rs += __shfl_xor(rs, off, 32);
            l[r] = l[r] * corr + rs;
            m[r] = mn;
            #pragma unroll
            for (int dc = 0; dc < 4; ++dc) acc[dc][r] *= corr;
            p0[r] = e0; p1[r] = e1;
        }

        // ---- transpose P via LDS (C-layout -> A-layout) ----
        #pragma unroll
        for (int r = 0; r < 8; ++r) {
            int rr = r + 8 * hg;
            ldsP[rr][ln]      = f2bf(p0[r]);
            ldsP[rr][16 + ln] = f2bf(p1[r]);
        }
        asm volatile("s_wait_dscnt 0" ::: "memory");

        BF16x16 pa;
        pa.q[0] = *(const v4u*)&ldsP[ln][hg * 8];        // ds_load_b128
        pa.q[1] = *(const v4u*)&ldsP[ln][16 + hg * 8];

        // ---- O += P V (V^T layout: key pairs contiguous) ----
        #pragma unroll
        for (int dc = 0; dc < 4; ++dc) {
            BF16x16 vb;
            const unsigned short* vcol = Vb + (size_t)(dc * 16 + ln) * TT + s0;
            vb.q[0] = *(const v4u*)(vcol + hg * 8);
            vb.q[1] = *(const v4u*)(vcol + 16 + hg * 8);
            acc[dc] = wmma_bf16(pa.v, vb.v, acc[dc]);
        }
    }

    // ---- normalize and store bf16 attention output [B,T,C] ----
    #pragma unroll
    for (int r = 0; r < 8; ++r) {
        float inv = 1.0f / l[r];
        int t = qBase + r + 8 * hg;
        #pragma unroll
        for (int dc = 0; dc < 4; ++dc) {
            int d = dc * 16 + ln;
            attn[((size_t)(b * TT + t)) * CC + h * DD + d] = f2bf(acc[dc][r] * inv);
        }
    }
}

// ===================================================================
// Kernel 3: out = attn @ Wout_t^T + b_out (fp32 result). 16x64 tile per wave.
// ===================================================================
__global__ void out_gemm_kernel(const unsigned short* __restrict__ attn,
                                const unsigned short* __restrict__ Wt,  // [CC, CC]
                                const float* __restrict__ bout,
                                float* __restrict__ out) {
    const int wave = (int)((blockIdx.x * blockDim.x + threadIdx.x) >> 5);
    const int lane = threadIdx.x & 31;
    const int tilesN = CC / 64;                  // 16
    const int tileN = (wave % tilesN) * 64;
    const int tileM = (wave / tilesN) * 16;
    if (tileM >= MROWS) return;
    const int hg = lane >> 4;
    const int ln = lane & 15;

    v8f acc[4];
    #pragma unroll
    for (int s = 0; s < 4; ++s) acc[s] = (v8f){0.f,0.f,0.f,0.f,0.f,0.f,0.f,0.f};

    const unsigned short* arow = attn + (size_t)(tileM + ln) * CC;
    const unsigned short* brow[4];
    #pragma unroll
    for (int s = 0; s < 4; ++s)
        brow[s] = Wt + (size_t)(tileN + s * 16 + ln) * CC;

    for (int k0 = 0; k0 < CC; k0 += 32) {
        BF16x16 a;
        a.q[0] = *(const v4u*)(arow + k0 + hg * 8);
        a.q[1] = *(const v4u*)(arow + k0 + 16 + hg * 8);
        #pragma unroll
        for (int s = 0; s < 4; ++s) {
            BF16x16 b;
            b.q[0] = *(const v4u*)(brow[s] + k0 + hg * 8);
            b.q[1] = *(const v4u*)(brow[s] + k0 + 16 + hg * 8);
            acc[s] = wmma_bf16(a.v, b.v, acc[s]);
        }
    }

    #pragma unroll
    for (int s = 0; s < 4; ++s) {
        int coln = tileN + s * 16 + ln;
        float bias = bout[coln];
        #pragma unroll
        for (int r = 0; r < 8; ++r) {
            int row = tileM + r + 8 * hg;
            out[(size_t)row * CC + coln] = acc[s][r] + bias;
        }
    }
}

// ===================================================================
extern "C" void kernel_launch(void* const* d_in, const int* in_sizes, int n_in,
                              void* d_out, int out_size, void* d_ws, size_t ws_size,
                              hipStream_t stream) {
    const float* x     = (const float*)d_in[0];
    const float* W_qkv = (const float*)d_in[1];
    const float* b_qkv = (const float*)d_in[2];
    const float* W_out = (const float*)d_in[3];
    const float* b_out = (const float*)d_in[4];
    float* out = (float*)d_out;

    // Workspace layout (bf16 elements):
    //   Q | K | V^T | attn | xb | Wqkv_t | Wout_t   = 48 MB total
    const size_t QKV_ELEMS = (size_t)BB * HH * TT * DD;      // 4,194,304
    unsigned short* Q      = (unsigned short*)d_ws;
    unsigned short* Kbuf   = Q      + QKV_ELEMS;
    unsigned short* Vt     = Kbuf   + QKV_ELEMS;
    unsigned short* attn   = Vt     + QKV_ELEMS;
    unsigned short* xb     = attn   + QKV_ELEMS;                 // 4096*1024
    unsigned short* Wqkv_t = xb     + (size_t)MROWS * CC;        // 3072*1024
    unsigned short* Wout_t = Wqkv_t + (size_t)N3 * CC;           // 1024*1024

    (void)in_sizes; (void)n_in; (void)out_size; (void)ws_size;

    // Pack pass: fp32 -> bf16 (x row-major; weights transposed).
    cvt_x_kernel <<<(MROWS * CC / 4) / 256, 256, 0, stream>>>(x, xb);
    cvt_wt_kernel<<<(N3 * CC / 4) / 256, 256, 0, stream>>>(W_qkv, Wqkv_t, CC, N3);
    cvt_wt_kernel<<<(CC * CC / 4) / 256, 256, 0, stream>>>(W_out, Wout_t, CC, CC);

    // QKV GEMM: 256 x 48 = 12288 wave-tiles (16x64 each), 8 waves/block.
    {
        int tiles = (MROWS / 16) * (N3 / 64);
        qkv_gemm_kernel<<<tiles / 8, 256, 0, stream>>>(xb, Wqkv_t, b_qkv, Q, Kbuf, Vt);
    }
    // Flash attention: 4096 single-wave blocks.
    flash_attn_kernel<<<BB * HH * (TT / 16), 32, 0, stream>>>(Q, Kbuf, Vt, attn);
    // Out GEMM: 256 x 16 = 4096 wave-tiles (16x64 each), 8 waves/block.
    {
        int tiles = (MROWS / 16) * (CC / 64);
        out_gemm_kernel<<<tiles / 8, 256, 0, stream>>>(attn, Wout_t, b_out, out);
    }
}
